// MultiHeadAttention_65017214927599
// MI455X (gfx1250) — matrix operand order repaired and checked by reading, not compiled, not run on hardware
//
#include <hip/hip_runtime.h>
#include <hip/hip_bf16.h>

#define B_ 4
#define S_ 2048
#define D_ 1024
#define H_ 16
#define DH_ 64

typedef __attribute__((ext_vector_type(16))) __bf16 v16bf;
typedef __attribute__((ext_vector_type(8)))  __bf16 bf16x8;
typedef __attribute__((ext_vector_type(8)))  float  v8f;

__device__ __forceinline__ unsigned short f2bf_bits(float f) {
    unsigned int u = __float_as_uint(f);
    return (unsigned short)((u + 0x7fffu + ((u >> 16) & 1u)) >> 16);
}
__device__ __forceinline__ __bf16 bfb(float f) {
    return __builtin_bit_cast(__bf16, f2bf_bits(f));
}

__device__ __forceinline__ v8f wmma_bf16(v16bf a, v16bf b, v8f c) {
    return __builtin_amdgcn_wmma_f32_16x16x32_bf16(
        /*neg_a=*/false, a, /*neg_b=*/false, b,
        /*c_mod=*/(short)0, c, /*reuse_a=*/false, /*reuse_b=*/false);
}

// A-fragment (16x32 bf16) from a row-major bf16 row (global or LDS).
// Lane layout: M = lane&15; half = lane>>4; K = k0 + half*16 .. +16 contiguous.
__device__ __forceinline__ v16bf loadA_bf(const __bf16* row, int k0, int half) {
    bf16x8 c0 = *(const bf16x8*)(row + k0 + half * 16);
    bf16x8 c1 = *(const bf16x8*)(row + k0 + half * 16 + 8);
    return __builtin_shufflevector(c0, c1, 0,1,2,3,4,5,6,7,8,9,10,11,12,13,14,15);
}
// B-fragment (32x16 bf16) from B^T rows (row-major [N,K]):
// lane N = lane&15; K = k0 + half*16 .. +16 contiguous (one 32B load).
__device__ __forceinline__ v16bf loadB_bf(const __bf16* rowT, int k0, int half) {
    return *(const v16bf*)(rowT + k0 + half * 16);
}

// ---------------------------------------------------------------------------
// Kernel 0: f32 -> bf16 bulk conversion (bandwidth-bound, one pass).
// Each thread: 8 elements (two float4 loads -> one 16B bf16x8 store).
// ---------------------------------------------------------------------------
__global__ __launch_bounds__(256) void cvt_kernel(
    const float* __restrict__ src, __bf16* __restrict__ dst)
{
    const int t = blockIdx.x * 256 + threadIdx.x;
    const float4* p = (const float4*)src + (size_t)t * 2;
    const float4 a = p[0], b = p[1];
    bf16x8 o;
    o[0] = bfb(a.x); o[1] = bfb(a.y); o[2] = bfb(a.z); o[3] = bfb(a.w);
    o[4] = bfb(b.x); o[5] = bfb(b.y); o[6] = bfb(b.z); o[7] = bfb(b.w);
    *(bf16x8*)(dst + (size_t)t * 8) = o;
}

// ---------------------------------------------------------------------------
// Kernel 1: projections (all-bf16 operands now).
// grid.z: 0 -> Q = query@Wq^T+bq  ([B,H,S,DH]); 1 -> K = key@Wq^T+bq (source
// bug: Wq) ([B,H,S,DH]); 2 -> V^T = (value@Wv^T+bv)^T ([B,H,DH,S]).
// Each wave computes a 64x64 tile of the 8192x1024 GEMM.
// ---------------------------------------------------------------------------
__global__ __launch_bounds__(256) void proj_kernel(
    const __bf16* __restrict__ inq, const __bf16* __restrict__ ink,
    const __bf16* __restrict__ inv,
    const __bf16* __restrict__ wqb, const __bf16* __restrict__ wvb,
    const float* __restrict__ bq, const float* __restrict__ bv,
    __bf16* __restrict__ Qbf, __bf16* __restrict__ Kbf, __bf16* __restrict__ Vt)
{
    const int z = blockIdx.z;
    const __bf16* A   = (z == 0) ? inq : (z == 1) ? ink : inv;
    const __bf16* W   = (z == 2) ? wvb : wqb;
    const float*  bia = (z == 2) ? bv : bq;
    __bf16* out       = (z == 0) ? Qbf : (z == 1) ? Kbf : Vt;
    const int layout  = (z == 2) ? 1 : 0;

    const int lane = threadIdx.x & 31;
    const int wave = threadIdx.x >> 5;
    const int half = lane >> 4;
    const int l15  = lane & 15;
    const int tile = blockIdx.x * 8 + wave;      // 2048 tiles = 128 (M) x 16 (N)
    const int mbase = (tile >> 4) * 64;
    const int nbase = (tile & 15) * 64;

    v8f acc[4][4];
#pragma unroll
    for (int i = 0; i < 4; i++)
#pragma unroll
        for (int j = 0; j < 4; j++) acc[i][j] = {};

    for (int k0 = 0; k0 < D_; k0 += 32) {
        v16bf af[4];
#pragma unroll
        for (int mi = 0; mi < 4; mi++)
            af[mi] = loadA_bf(A + (size_t)(mbase + mi * 16 + l15) * D_, k0, half);
#pragma unroll
        for (int ni = 0; ni < 4; ni++) {
            const v16bf bf = loadB_bf(W + (size_t)(nbase + ni * 16 + l15) * D_, k0, half);
#pragma unroll
            for (int mi = 0; mi < 4; mi++)
                acc[mi][ni] = wmma_bf16(af[mi], bf, acc[mi][ni]);
        }
    }

#pragma unroll
    for (int mi = 0; mi < 4; mi++)
#pragma unroll
        for (int ni = 0; ni < 4; ni++) {
            const int n = nbase + ni * 16 + l15;
            const float bias = bia[n];
#pragma unroll
            for (int r = 0; r < 8; r++) {
                const int m = mbase + mi * 16 + 8 * half + r;
                const float v = acc[mi][ni][r] + bias;
                const int b = m >> 11, s = m & (S_ - 1);
                const int h = n >> 6,  dh = n & (DH_ - 1);
                size_t idx;
                if (layout == 0)
                    idx = (((size_t)b * H_ + h) * S_ + s) * DH_ + dh;
                else
                    idx = (((size_t)b * H_ + h) * DH_ + dh) * S_ + s;
                out[idx] = bfb(v);
            }
        }
}

// ---------------------------------------------------------------------------
// Kernel 2: attention. One block = (b,h, 16 query rows).
// Phase 1: scores (scaled) -> LDS f32. Phase 2: coalesced mask apply + exact
// softmax; probs written to HBM (f32) and to LDS as bf16. Phase 3: ctx = P@V
// with direct bf16 LDS A-fragments, per-wave K-split, LDS reduction.
// LDS: sc[16][2048] f32 | red[16][17] | rowbc[16] | cpart[8][16][64] | pb[16][2048] bf16
// ---------------------------------------------------------------------------
#define SMEM_A (((16 * 2048 + 16 * 17 + 16 + 8 * 16 * 64) * 4) + (16 * 2048 * 2))

__global__ __launch_bounds__(256) void attn_kernel(
    const __bf16* __restrict__ Qbf, const __bf16* __restrict__ Kbf,
    const __bf16* __restrict__ Vt, const int* __restrict__ mask,
    float* __restrict__ probs, __bf16* __restrict__ ctxbf)
{
    extern __shared__ float smem[];
    float* sc    = smem;                   // 16 x 2048 scores
    float* red   = sc + 16 * 2048;         // [16][17] reduction scratch
    float* rowbc = red + 16 * 17;          // [16] broadcast
    float* cpart = rowbc + 16;             // [8][16][64] partial ctx
    __bf16* pb   = (__bf16*)(cpart + 8 * 16 * 64);  // 16 x 2048 probs (bf16)

    const int blk = blockIdx.x;            // (b*H + h)*128 + qt
    const int qt  = blk & 127;
    const int bh  = blk >> 7;
    const int b   = bh >> 4;
    const int h   = bh & 15;

    const int lane = threadIdx.x & 31;
    const int wave = threadIdx.x >> 5;
    const int half = lane >> 4;
    const int l15  = lane & 15;

    // ---- Phase 1: scores = (q k^T)/8 into LDS ----
    const __bf16* qrow = Qbf + ((size_t)bh * S_ + qt * 16 + l15) * DH_;
    const v16bf a0 = loadA_bf(qrow, 0, half);
    const v16bf a1 = loadA_bf(qrow, 32, half);

    const size_t kbase = (size_t)bh * S_;
    for (int nt = 0; nt < 16; nt++) {
        const int n0 = wave * 256 + nt * 16;
        const __bf16* krow = Kbf + (kbase + n0 + l15) * DH_;
        const v16bf b0 = loadB_bf(krow, 0, half);
        const v16bf b1 = loadB_bf(krow, 32, half);
        v8f acc = {};
        acc = wmma_bf16(a0, b0, acc);
        acc = wmma_bf16(a1, b1, acc);
        const int col = n0 + l15;
#pragma unroll
        for (int r = 0; r < 8; r++)
            sc[(8 * half + r) * S_ + col] = acc[r] * 0.125f;   // 1/sqrt(64)
    }
    __syncthreads();

    // ---- Phase 2: mask (coalesced) + softmax; probs -> HBM(f32) + LDS(bf16) ----
    const int row = threadIdx.x >> 4;
    const int sub = threadIdx.x & 15;
    const int qs  = qt * 16 + row;
    const int* mrow = mask + ((size_t)b * S_ + qs) * S_;

    float mx = -3.4e38f;
    for (int j = 0; j < 128; j++) {
        const int c = sub + j * 16;
        float v = sc[row * S_ + c];
        if (mrow[c] != 0) v = -1e9f;
        sc[row * S_ + c] = v;
        mx = fmaxf(mx, v);
    }
    red[row * 17 + sub] = mx;
    __syncthreads();
    if (sub == 0) {
        float m2 = red[row * 17];
        for (int i = 1; i < 16; i++) m2 = fmaxf(m2, red[row * 17 + i]);
        rowbc[row] = m2;
    }
    __syncthreads();
    const float rmax = rowbc[row];
    float psum = 0.f;
    for (int j = 0; j < 128; j++) {
        const int c = sub + j * 16;
        const float e = __expf(sc[row * S_ + c] - rmax);
        sc[row * S_ + c] = e;
        psum += e;
    }
    red[row * 17 + sub] = psum;
    __syncthreads();
    if (sub == 0) {
        float s2 = 0.f;
        for (int i = 0; i < 16; i++) s2 += red[row * 17 + i];
        rowbc[row] = 1.0f / s2;
    }
    __syncthreads();
    const float inv = rowbc[row];
    const size_t pbase = ((size_t)bh * S_ + qs) * S_;
    for (int j = 0; j < 128; j++) {
        const int c = sub + j * 16;
        const float pv = sc[row * S_ + c] * inv;
        probs[pbase + c] = pv;
        pb[row * S_ + c] = bfb(pv);
    }
    __syncthreads();

    // ---- Phase 3: ctx = P @ V  (wave w handles K in [w*256, +256)) ----
    v8f acc4[4];
#pragma unroll
    for (int i = 0; i < 4; i++) acc4[i] = {};
    const size_t vtbase = (size_t)bh * DH_;
    const __bf16* prow = pb + l15 * S_;
    for (int kk = 0; kk < 8; kk++) {
        const int k0 = wave * 256 + kk * 32;
        const v16bf af = loadA_bf(prow, k0, half);
#pragma unroll
        for (int ni = 0; ni < 4; ni++) {
            const __bf16* vrow = Vt + (vtbase + ni * 16 + l15) * S_;
            acc4[ni] = wmma_bf16(af, loadB_bf(vrow, k0, half), acc4[ni]);
        }
    }
#pragma unroll
    for (int ni = 0; ni < 4; ni++)
#pragma unroll
        for (int r = 0; r < 8; r++)
            cpart[wave * 1024 + (8 * half + r) * 64 + ni * 16 + l15] = acc4[ni][r];
    __syncthreads();
#pragma unroll
    for (int i = 0; i < 4; i++) {
        const int o = threadIdx.x * 4 + i;       // 0..1023 = m*64 + d
        float s = 0.f;
#pragma unroll
        for (int w = 0; w < 8; w++) s += cpart[w * 1024 + o];
        const int m = o >> 6, d = o & 63;
        ctxbf[((size_t)b * S_ + qt * 16 + m) * D_ + h * DH_ + d] = bfb(s);
    }
}

// ---------------------------------------------------------------------------
// Kernel 3: out = LayerNorm(ctx @ Wo^T + bo + residual) * gamma + beta.
// Block = 16 rows; wave w covers N in [w*128, +128). Row in LDS for LN.
// ---------------------------------------------------------------------------
#define SMEM_O ((16 * 1024 + 16 * 17 + 16 + 16) * 4)

__global__ __launch_bounds__(256) void outln_kernel(
    const __bf16* __restrict__ ctxbf, const __bf16* __restrict__ wob,
    const float* __restrict__ bo, const float* __restrict__ query,
    const float* __restrict__ gamma, const float* __restrict__ beta,
    float* __restrict__ out)
{
    extern __shared__ float smem[];
    float* xs     = smem;              // [16][1024]
    float* red    = xs + 16 * 1024;    // [16][17]
    float* rowmu  = red + 16 * 17;     // [16]
    float* rowist = rowmu + 16;        // [16]

    const int rbase = blockIdx.x * 16;
    const int lane = threadIdx.x & 31;
    const int wave = threadIdx.x >> 5;
    const int half = lane >> 4;
    const int l15  = lane & 15;

    v8f acc[8];
#pragma unroll
    for (int i = 0; i < 8; i++) acc[i] = {};

    const __bf16* arow = ctxbf + (size_t)(rbase + l15) * D_;
    for (int k0 = 0; k0 < D_; k0 += 32) {
        const v16bf af = loadA_bf(arow, k0, half);
#pragma unroll
        for (int nt = 0; nt < 8; nt++) {
            const __bf16* wrow = wob + (size_t)(wave * 128 + nt * 16 + l15) * D_;
            acc[nt] = wmma_bf16(af, loadB_bf(wrow, k0, half), acc[nt]);
        }
    }

#pragma unroll
    for (int nt = 0; nt < 8; nt++) {
        const int n = wave * 128 + nt * 16 + l15;
        const float bias = bo[n];
#pragma unroll
        for (int r = 0; r < 8; r++) {
            const int m = 8 * half + r;
            const int rowg = rbase + m;
            xs[m * D_ + n] = acc[nt][r] + bias + query[(size_t)rowg * D_ + n];
        }
    }
    __syncthreads();

    const int row = threadIdx.x >> 4;
    const int sub = threadIdx.x & 15;
    float s = 0.f, s2 = 0.f;
    for (int j = 0; j < 64; j++) {
        const float x = xs[row * D_ + sub + j * 16];
        s += x; s2 += x * x;
    }
    red[row * 17 + sub] = s;
    __syncthreads();
    if (sub == 0) {
        float t = 0.f;
        for (int i = 0; i < 16; i++) t += red[row * 17 + i];
        rowmu[row] = t * (1.0f / D_);
    }
    __syncthreads();
    const float mu = rowmu[row];
    red[row * 17 + sub] = s2;
    __syncthreads();
    if (sub == 0) {
        float t = 0.f;
        for (int i = 0; i < 16; i++) t += red[row * 17 + i];
        const float var = t * (1.0f / D_) - mu * mu;
        rowist[row] = rsqrtf(var + 1e-12f);
    }
    __syncthreads();
    const float istd = rowist[row];
    const size_t obase = (size_t)(rbase + row) * D_;
    for (int j = 0; j < 64; j++) {
        const int c = sub + j * 16;
        out[obase + c] = (xs[row * D_ + c] - mu) * istd * gamma[c] + beta[c];
    }
}

extern "C" void kernel_launch(void* const* d_in, const int* in_sizes, int n_in,
                              void* d_out, int out_size, void* d_ws, size_t ws_size,
                              hipStream_t stream) {
    (void)in_sizes; (void)n_in; (void)out_size; (void)ws_size;

    const float* query = (const float*)d_in[0];
    const float* key   = (const float*)d_in[1];
    const float* value = (const float*)d_in[2];
    const int*   mask  = (const int*)d_in[3];
    const float* Wq    = (const float*)d_in[4];
    const float* bq    = (const float*)d_in[5];
    const float* Wv    = (const float*)d_in[6];
    const float* bv    = (const float*)d_in[7];
    const float* Wo    = (const float*)d_in[8];
    const float* bo    = (const float*)d_in[9];
    const float* gamma = (const float*)d_in[10];
    const float* beta  = (const float*)d_in[11];

    float* out   = (float*)d_out;
    float* probs = out + (size_t)B_ * S_ * D_;      // tuple: (out, probs)

    const size_t NA = (size_t)B_ * S_ * D_;         // 8,388,608 activation elems
    const size_t NW = (size_t)D_ * D_;              // 1,048,576 weight elems

    __bf16* Qbf  = (__bf16*)d_ws;                   // 16 MB each
    __bf16* Kbf  = Qbf + NA;
    __bf16* Vt   = Kbf + NA;
    __bf16* inq  = Vt  + NA;                        // bf16 activations (48 MB)
    __bf16* ink  = inq + NA;
    __bf16* invv = ink + NA;
    __bf16* wqb  = invv + NA;                       // bf16 weights (6 MB)
    __bf16* wvb  = wqb + NW;
    __bf16* wob  = wvb + NW;
    __bf16* ctxbf = inq;   // alias: bf16 inputs dead after proj_kernel

    hipFuncSetAttribute(reinterpret_cast<const void*>(attn_kernel),
                        hipFuncAttributeMaxDynamicSharedMemorySize, SMEM_A);
    hipFuncSetAttribute(reinterpret_cast<const void*>(outln_kernel),
                        hipFuncAttributeMaxDynamicSharedMemorySize, SMEM_O);

    // f32 -> bf16 staging (one pass each; bandwidth-bound)
    cvt_kernel<<<dim3(NA / (256 * 8)), 256, 0, stream>>>(query, inq);
    cvt_kernel<<<dim3(NA / (256 * 8)), 256, 0, stream>>>(key,   ink);
    cvt_kernel<<<dim3(NA / (256 * 8)), 256, 0, stream>>>(value, invv);
    cvt_kernel<<<dim3(NW / (256 * 8)), 256, 0, stream>>>(Wq, wqb);
    cvt_kernel<<<dim3(NW / (256 * 8)), 256, 0, stream>>>(Wv, wvb);
    cvt_kernel<<<dim3(NW / (256 * 8)), 256, 0, stream>>>(Wo, wob);

    proj_kernel<<<dim3(256, 1, 3), 256, 0, stream>>>(
        inq, ink, invv, wqb, wvb, bq, bv, Qbf, Kbf, Vt);

    attn_kernel<<<dim3(B_ * H_ * (S_ / 16)), 256, SMEM_A, stream>>>(
        Qbf, Kbf, Vt, mask, probs, ctxbf);

    outln_kernel<<<dim3((B_ * S_) / 16), 256, SMEM_O, stream>>>(
        ctxbf, wob, bo, query, gamma, beta, out);
}
